// ScaledDotProductAttention_36593121362165
// MI455X (gfx1250) — matrix operand order repaired and checked by reading, not compiled, not run on hardware
//
#include <hip/hip_runtime.h>
#include <hip/hip_bf16.h>
#include <stdint.h>

// Problem constants (match reference)
#define B_   2
#define W_   2048
#define C_   1024
#define H_   16
#define KD   64          // head dim
#define BW_  (B_ * W_)   // 4096 rows

typedef __attribute__((ext_vector_type(16))) __bf16 v16bf;
typedef __attribute__((ext_vector_type(8)))  float  v8f;

union Frag16 { uint4 u[2]; v16bf v; };

__device__ __forceinline__ unsigned short f2bf(float f) {
    unsigned int u = __float_as_uint(f);
    unsigned int r = u + 0x7FFFu + ((u >> 16) & 1u);   // round-to-nearest-even
    return (unsigned short)(r >> 16);
}

__device__ __forceinline__ v8f wmma_bf16(v16bf a, v16bf b, v8f c) {
    // D = A(16x32 bf16) * B(32x16 bf16) + C(16x16 f32)
    return __builtin_amdgcn_wmma_f32_16x16x32_bf16(false, a, false, b,
                                                   (short)0, c, false, false);
}

// CDNA5 async global->LDS copy (GLOBAL_LOAD_ASYNC_TO_LDS_B128, ASYNCcnt).
// GVS addressing: mem = SGPR_base + VGPR_i32_offset; vdst = LDS byte offset.
__device__ __forceinline__ void async_load_b128(unsigned lds_off, const void* base,
                                                int voff_bytes) {
    asm volatile("global_load_async_to_lds_b128 %0, %1, %2"
                 :: "v"(lds_off), "v"(voff_bytes), "s"(base)
                 : "memory");
}

__device__ __forceinline__ void wait_async0() {
    asm volatile("s_wait_asynccnt 0x0" ::: "memory");
}

// LDS byte offset of a __shared__ pointer: flat addr[31:0] is the wave-relative
// LDS address (ISA 10.2 aperture mapping).
__device__ __forceinline__ unsigned lds_off(const void* p) {
    return (unsigned)(uintptr_t)p;
}

// ---------------------------------------------------------------------------
// Kernel 0: LayerNorm both sequences, emit bf16. One wave (32 lanes) per row.
// ---------------------------------------------------------------------------
__global__ __launch_bounds__(256)
void ln_kernel(const float* __restrict__ seq1, const float* __restrict__ seq2,
               const float* __restrict__ gamma, const float* __restrict__ beta,
               unsigned short* __restrict__ x1, unsigned short* __restrict__ x2) {
    int wave = threadIdx.x >> 5, lane = threadIdx.x & 31;
    int row  = blockIdx.x * 8 + wave;                 // 0 .. 2*BW_-1
    const float* src;
    unsigned short* dst;
    if (row < BW_) { src = seq1 + (size_t)row * C_;        dst = x1 + (size_t)row * C_; }
    else           { src = seq2 + (size_t)(row - BW_) * C_; dst = x2 + (size_t)(row - BW_) * C_; }

    float4 xv[8];
    float s = 0.f, sq = 0.f;
    for (int i = 0; i < 8; ++i) {
        xv[i] = *(const float4*)(src + i * 128 + lane * 4);
        s  += xv[i].x + xv[i].y + xv[i].z + xv[i].w;
        sq += xv[i].x * xv[i].x + xv[i].y * xv[i].y + xv[i].z * xv[i].z + xv[i].w * xv[i].w;
    }
    for (int m = 16; m >= 1; m >>= 1) {
        s  += __shfl_xor(s,  m, 32);
        sq += __shfl_xor(sq, m, 32);
    }
    float mu   = s * (1.f / C_);
    float var  = sq * (1.f / C_) - mu * mu;
    float rstd = rsqrtf(var + 1e-5f);

    for (int i = 0; i < 8; ++i) {
        int c = i * 128 + lane * 4;
        float y0 = (xv[i].x - mu) * rstd * gamma[c + 0] + beta[c + 0];
        float y1 = (xv[i].y - mu) * rstd * gamma[c + 1] + beta[c + 1];
        float y2 = (xv[i].z - mu) * rstd * gamma[c + 2] + beta[c + 2];
        float y3 = (xv[i].w - mu) * rstd * gamma[c + 3] + beta[c + 3];
        uint2 o;
        o.x = (unsigned)f2bf(y0) | ((unsigned)f2bf(y1) << 16);
        o.y = (unsigned)f2bf(y2) | ((unsigned)f2bf(y3) << 16);
        *(uint2*)(dst + c) = o;
    }
}

// ---------------------------------------------------------------------------
// Kernel 1: QKV projection GEMM. grid.z selects Q/K/V. Block = 4 waves,
// 64(M) x 64(N) tile, k-loop over C in steps of 32, WMMA bf16.
// Q,V stored [B,H,W,KD]; K stored transposed [B,H,KD,W] for attention B-frags.
// ---------------------------------------------------------------------------
__global__ __launch_bounds__(128)
void qkv_kernel(const unsigned short* __restrict__ x1,
                const unsigned short* __restrict__ x2,
                const float* __restrict__ qw, const float* __restrict__ kw,
                const float* __restrict__ vw,
                unsigned short* __restrict__ Qb,
                unsigned short* __restrict__ Ktb,
                unsigned short* __restrict__ Vb) {
    __shared__ unsigned short sA[64 * 32];   // [m][k] rows of 64B
    __shared__ unsigned short sB[32 * 64];   // [k][n] rows of 128B

    int sel = blockIdx.z;
    const unsigned short* X = (sel == 0) ? x1 : x2;
    const float* Wm = (sel == 0) ? qw : (sel == 1 ? kw : vw);

    int r0 = blockIdx.x * 64;      // row tile (b*W+w)
    int n0 = blockIdx.y * 64;      // col tile; 64-aligned -> single head
    int h  = n0 >> 6;
    int t = threadIdx.x, wave = t >> 5, lane = t & 31;
    int mlo = lane & 15, half = lane >> 4;
    unsigned ldsA = lds_off(sA);

    v8f acc[4] = {};

    for (int k0 = 0; k0 < C_; k0 += 32) {
        __syncthreads();
        // A tile: 64x32 bf16 (256 x 16B) via async global->LDS, 2 per thread
        for (int i = 0; i < 2; ++i) {
            int idx = t + i * 128;
            int rr = idx >> 2, cc = idx & 3;
            async_load_b128(ldsA + idx * 16, X,
                            ((r0 + rr) * C_ + k0 + cc * 8) * 2);
        }
        // B tile: 32(c) x 64(n=kf within head) f32 -> bf16 (VALU convert path)
        {
            int c  = k0 + (t >> 2);
            int nb = (t & 3) * 16;
            const float* src = Wm + ((size_t)h * C_ + c) * KD + nb;
            unsigned short* d = sB + (size_t)(t >> 2) * 64 + nb;
            for (int j = 0; j < 16; ++j) d[j] = f2bf(src[j]);
        }
        wait_async0();
        __syncthreads();

        Frag16 a;
        const uint4* ar = (const uint4*)(sA + (wave * 16 + mlo) * 32);
        a.u[0] = ar[half];
        a.u[1] = ar[2 + half];
        for (int nt = 0; nt < 4; ++nt) {
            Frag16 b;
            const unsigned short* bc = sB + nt * 16 + mlo;
            for (int s = 0; s < 16; ++s) ((unsigned short*)&b)[s] = bc[((s & 7) + 8 * half + 16 * (s >> 3)) * 64];
            acc[nt] = wmma_bf16(a.v, b.v, acc[nt]);
        }
    }

    for (int nt = 0; nt < 4; ++nt) {
        for (int r = 0; r < 8; ++r) {
            int row = r0 + wave * 16 + half * 8 + r;
            int b   = row >> 11;            // / W_
            int w   = row & (W_ - 1);
            int kf  = (n0 + nt * 16 + mlo) & 63;
            unsigned short val = f2bf(acc[nt][r]);
            if (sel == 1) {
                Ktb[(((size_t)b * H_ + h) * KD + kf) * W_ + w] = val;
            } else {
                unsigned short* o = (sel == 0) ? Qb : Vb;
                o[(((size_t)b * H_ + h) * W_ + w) * KD + kf] = val;
            }
        }
    }
}

// ---------------------------------------------------------------------------
// Kernel 2: causal flash attention per (b,h). Block = 4 waves = 64 queries.
// Key blocks of 64, online softmax in f32 D-fragments, P restaged via LDS.
// K/V tiles staged with GLOBAL_LOAD_ASYNC_TO_LDS_B128.
// ---------------------------------------------------------------------------
__global__ __launch_bounds__(128)
void attn_kernel(const unsigned short* __restrict__ Qb,
                 const unsigned short* __restrict__ Ktb,
                 const unsigned short* __restrict__ Vb,
                 unsigned short* __restrict__ Ob) {
    __shared__ unsigned short sK[64 * 64];      // [kdim][key] rows of 128B
    __shared__ unsigned short sV[64 * 64];      // [key][feat] rows of 128B
    __shared__ unsigned short sP[4][16 * 64];   // per-wave probs [m][key]

    int bh = blockIdx.y;
    int b = bh >> 4, h = bh & 15;
    int qblk = blockIdx.x;
    int q0 = qblk * 64;
    int t = threadIdx.x, wave = t >> 5, lane = t & 31;
    int mlo = lane & 15, half = lane >> 4;

    const unsigned short* Qp  = Qb  + (((size_t)b * H_ + h) * W_) * KD;
    const unsigned short* Ktp = Ktb + (((size_t)b * H_ + h) * KD) * W_;
    const unsigned short* Vp  = Vb  + (((size_t)b * H_ + h) * W_) * KD;
    unsigned ldsK = lds_off(sK), ldsV = lds_off(sV);

    // Q fragments for this wave's 16 rows (kdim 0..63 -> 2 A-frags), kept live
    Frag16 qf[2];
    {
        const uint4* qr = (const uint4*)(Qp + (size_t)(q0 + wave * 16 + mlo) * KD);
        qf[0].u[0] = qr[half];     qf[0].u[1] = qr[2 + half];
        qf[1].u[0] = qr[4 + half]; qf[1].u[1] = qr[6 + half];
    }

    v8f oacc[4] = {};
    float mrow[8], lrow[8];
    for (int r = 0; r < 8; ++r) { mrow[r] = -3.0e38f; lrow[r] = 0.f; }
    const float scale = 0.125f;   // 1/sqrt(64)

    for (int kb = 0; kb <= qblk; ++kb) {
        __syncthreads();
        // async-stage K^T tile ([kdim 0..63][keys kb*64..+64]) and V tile
        for (int i = 0; i < 4; ++i) {
            int idx = t + i * 128;
            int rr = idx >> 3, cc = idx & 7;
            async_load_b128(ldsK + idx * 16, Ktp,
                            (rr * W_ + kb * 64 + cc * 8) * 2);
            async_load_b128(ldsV + idx * 16, Vp,
                            ((kb * 64 + rr) * KD + cc * 8) * 2);
        }
        wait_async0();
        __syncthreads();

        // scores S = Q * K^T for 4 key sub-tiles of 16
        v8f s[4];
        for (int nt = 0; nt < 4; ++nt) {
            v8f sc = {};
            for (int kt = 0; kt < 2; ++kt) {
                Frag16 bk;
                const unsigned short* bc = sK + (kt * 32) * 64 + nt * 16 + mlo;
                for (int s = 0; s < 16; ++s) ((unsigned short*)&bk)[s] = bc[((s & 7) + 8 * half + 16 * (s >> 3)) * 64];
                sc = wmma_bf16(qf[kt].v, bk.v, sc);
            }
            s[nt] = sc;
        }

        // online softmax (row stats live per D-frag row r in each 16-lane half)
        bool diag = (kb == qblk);
        for (int r = 0; r < 8; ++r) {
            int mg = q0 + wave * 16 + half * 8 + r;
            float mx = -3.0e38f;
            for (int nt = 0; nt < 4; ++nt) {
                float v = s[nt][r] * scale;
                if (diag) {
                    int ng = kb * 64 + nt * 16 + mlo;
                    if (ng > mg) v = -3.0e38f;
                }
                s[nt][r] = v;
                mx = fmaxf(mx, v);
            }
            for (int xm = 1; xm <= 8; xm <<= 1) mx = fmaxf(mx, __shfl_xor(mx, xm, 32));
            float mn = fmaxf(mrow[r], mx);
            float alpha = __expf(mrow[r] - mn);
            float psum = 0.f;
            for (int nt = 0; nt < 4; ++nt) {
                float p = __expf(s[nt][r] - mn);
                s[nt][r] = p;
                psum += p;
            }
            for (int xm = 1; xm <= 8; xm <<= 1) psum += __shfl_xor(psum, xm, 32);
            lrow[r] = lrow[r] * alpha + psum;
            mrow[r] = mn;
            for (int nt = 0; nt < 4; ++nt) oacc[nt][r] *= alpha;
        }

        // restage P as bf16 A-fragments (wave-private LDS slab)
        unsigned short* sp = sP[wave];
        for (int nt = 0; nt < 4; ++nt)
            for (int r = 0; r < 8; ++r)
                sp[(half * 8 + r) * 64 + nt * 16 + mlo] = f2bf(s[nt][r]);
        asm volatile("s_wait_dscnt 0x0" ::: "memory");   // wave-internal LDS RAW

        // O += P * V
        for (int kt = 0; kt < 2; ++kt) {
            Frag16 ap;
            const uint4* ar = (const uint4*)(sp + mlo * 64 + kt * 32);
            ap.u[0] = ar[half]; ap.u[1] = ar[2 + half];
            for (int nt = 0; nt < 4; ++nt) {
                Frag16 bv;
                const unsigned short* bc = sV + (kt * 32) * 64 + nt * 16 + mlo;
                for (int s = 0; s < 16; ++s) ((unsigned short*)&bv)[s] = bc[((s & 7) + 8 * half + 16 * (s >> 3)) * 64];
                oacc[nt] = wmma_bf16(ap.v, bv.v, oacc[nt]);
            }
        }
    }

    // normalize + store interleaved as [B,W,C] with c = h*KD + feat
    for (int r = 0; r < 8; ++r) {
        float inv = 1.0f / lrow[r];
        int w = q0 + wave * 16 + half * 8 + r;
        for (int nt = 0; nt < 4; ++nt) {
            float v = oacc[nt][r] * inv;
            Ob[((size_t)b * W_ + w) * C_ + h * KD + nt * 16 + mlo] = f2bf(v);
        }
    }
}

// ---------------------------------------------------------------------------
// Kernel 3: mixer GEMM  out = O_bf16 @ W^T + bias  (f32 epilogue to d_out)
// ---------------------------------------------------------------------------
__global__ __launch_bounds__(128)
void mixer_kernel(const unsigned short* __restrict__ Ob,
                  const float* __restrict__ Wm,
                  const float* __restrict__ bias,
                  float* __restrict__ out) {
    __shared__ unsigned short sA[64 * 32];
    __shared__ unsigned short sB[32 * 64];

    int r0 = blockIdx.x * 64, n0 = blockIdx.y * 64;
    int t = threadIdx.x, wave = t >> 5, lane = t & 31;
    int mlo = lane & 15, half = lane >> 4;
    unsigned ldsA = lds_off(sA);

    v8f acc[4] = {};
    for (int k0 = 0; k0 < C_; k0 += 32) {
        __syncthreads();
        for (int i = 0; i < 2; ++i) {
            int idx = t + i * 128;
            int rr = idx >> 2, cc = idx & 3;
            async_load_b128(ldsA + idx * 16, Ob,
                            ((r0 + rr) * C_ + k0 + cc * 8) * 2);
        }
        {   // W[c][j] -> sB[j][c-n0]  (B-frag: row = kdim j, contiguous n)
            int n  = n0 + (t >> 1);
            int jb = (t & 1) * 16;
            const float* src = Wm + (size_t)n * C_ + k0 + jb;
            for (int j = 0; j < 16; ++j) sB[(jb + j) * 64 + (t >> 1)] = f2bf(src[j]);
        }
        wait_async0();
        __syncthreads();

        Frag16 a;
        const uint4* ar = (const uint4*)(sA + (wave * 16 + mlo) * 32);
        a.u[0] = ar[half]; a.u[1] = ar[2 + half];
        for (int nt = 0; nt < 4; ++nt) {
            Frag16 b;
            const unsigned short* bc = sB + nt * 16 + mlo;
            for (int s = 0; s < 16; ++s) ((unsigned short*)&b)[s] = bc[((s & 7) + 8 * half + 16 * (s >> 3)) * 64];
            acc[nt] = wmma_bf16(a.v, b.v, acc[nt]);
        }
    }

    for (int nt = 0; nt < 4; ++nt)
        for (int r = 0; r < 8; ++r) {
            int row = r0 + wave * 16 + half * 8 + r;
            int n   = n0 + nt * 16 + mlo;
            out[(size_t)row * C_ + n] = acc[nt][r] + bias[n];
        }
}

// ---------------------------------------------------------------------------
// Host-side launcher. Workspace layout (bf16 elements):
//   x1_ln, x2_ln, Q[B,H,W,KD], Kt[B,H,KD,W], V[B,H,W,KD], O[B,W,C]
//   = 6 * 4Mi bf16 = 48 MB total.
// ---------------------------------------------------------------------------
extern "C" void kernel_launch(void* const* d_in, const int* in_sizes, int n_in,
                              void* d_out, int out_size, void* d_ws, size_t ws_size,
                              hipStream_t stream) {
    const float* seq1  = (const float*)d_in[0];
    const float* seq2  = (const float*)d_in[1];
    const float* gamma = (const float*)d_in[2];
    const float* beta  = (const float*)d_in[3];
    const float* qw    = (const float*)d_in[4];
    const float* kw    = (const float*)d_in[5];
    const float* vw    = (const float*)d_in[6];
    const float* mw    = (const float*)d_in[7];
    const float* mb    = (const float*)d_in[8];

    const size_t N = (size_t)BW_ * C_;   // 4 Mi elements
    unsigned short* x1  = (unsigned short*)d_ws;
    unsigned short* x2  = x1  + N;
    unsigned short* Qb  = x2  + N;
    unsigned short* Ktb = Qb  + N;
    unsigned short* Vb  = Ktb + N;
    unsigned short* Ob  = Vb  + N;
    float* out = (float*)d_out;

    ln_kernel<<<dim3((2 * BW_) / 8), 256, 0, stream>>>(seq1, seq2, gamma, beta, x1, x2);
    qkv_kernel<<<dim3(BW_ / 64, (H_ * KD) / 64, 3), 128, 0, stream>>>(
        x1, x2, qw, kw, vw, Qb, Ktb, Vb);
    attn_kernel<<<dim3(W_ / 64, B_ * H_), 128, 0, stream>>>(Qb, Ktb, Vb, Ob);
    mixer_kernel<<<dim3(BW_ / 64, C_ / 64), 128, 0, stream>>>(Ob, mw, mb, out);
}
